// AttentionBlstmQuora_74964359185181
// MI455X (gfx1250) — matrix operand order repaired and checked by reading, not compiled, not run on hardware
//
#include <hip/hip_runtime.h>
#include <math.h>
#include <stdint.h>

typedef __attribute__((ext_vector_type(16))) _Float16 v16h;
typedef __attribute__((ext_vector_type(8)))  _Float16 v8h;
typedef __attribute__((ext_vector_type(8)))  float    v8f;
typedef unsigned int u32x4 __attribute__((ext_vector_type(4)));
typedef int          i32x4 __attribute__((ext_vector_type(4)));
typedef int          i32x8 __attribute__((ext_vector_type(8)));

#define WMMA16(a,b,c) __builtin_amdgcn_wmma_f32_16x16x32_f16(false,(a),false,(b),(short)0,(c),false,false)

#if defined(__gfx1250__) && __has_builtin(__builtin_amdgcn_tensor_load_to_lds) && \
    __has_builtin(__builtin_amdgcn_s_wait_tensorcnt)
#define USE_TDM 1
#else
#define USE_TDM 0
#endif

#define NB   64
#define NT   121
#define NR   7744      /* NB*NT */
#define NE   300
#define NEP  320
#define NH4  1024      /* 4*H */
#define ND   512
#define ND3  1536
#define ND4  2048

__device__ __forceinline__ float sigf(float x) { return 1.f / (1.f + __expf(-x)); }

// A fragment: 16x32 f16 tile, row-major, stride lda (elements).
// lane m=l&15, hi=l>>4; 16 contiguous f16 at K=hi*16..+16 (matches B fragment)
__device__ __forceinline__ v16h load_frag_a(const _Float16* base, int lda, int lane) {
  int m = lane & 15, hi = lane >> 4;
  const _Float16* p = base + (size_t)m * lda + hi * 16;
  v16h r;
  *(v8h*)&r       = *(const v8h*)p;
  *((v8h*)&r + 1) = *(const v8h*)(p + 8);
  return r;
}

// B fragment: 32x16 f16 tile stored N-major (stride ldb along K).
// lane n=l&15, hi=l>>4; 16 contiguous f16 at BT[n][hi*16] give V[i]=(K=hi*16+2i, +1)
__device__ __forceinline__ v16h load_frag_b(const _Float16* bt, int ldb, int lane) {
  int n = lane & 15, hi = lane >> 4;
  const _Float16* p = bt + (size_t)n * ldb + hi * 16;
  v16h r;
  *(v8h*)&r       = *(const v8h*)p;
  *((v8h*)&r + 1) = *(const v8h*)(p + 8);
  return r;
}

#if USE_TDM
// TDM descriptor (ISA cdna5 §8): load one 64-row x 32-col f16 tile (row stride
// `strideElems` f16) into LDS at byte offset lds_addr. The tile is presented as
// its own 64x32 tensor so no OOB path triggers. LDS padding: after every
// 16 DWORDs (=64B row) insert 4 DWORDs (=16B) -> LDS row stride 40 f16 (=LDT).
__device__ __forceinline__ void tdm_load_tile_64x32(const _Float16* gsrc,
                                                    uint32_t lds_addr,
                                                    uint32_t strideElems) {
  uint64_t ga = (uint64_t)(uintptr_t)gsrc;
  u32x4 g0;
  g0[0] = 1u;                                            // count=1 (valid descriptor)
  g0[1] = lds_addr;                                      // lds_addr [63:32]
  g0[2] = (uint32_t)ga;                                  // global_addr low
  g0[3] = (uint32_t)((ga >> 32) & 0x01FFFFFFull) | (2u << 30); // ga hi | type=2
  i32x8 g1;
  g1[0] = (1 << 16) | (1 << 20) | (3 << 22) | (3 << 25); // data_size=2B, pad_en, int=16DW, amt=4DW
  g1[1] = (32 << 16);                                    // tensor_dim0 = 32
  g1[2] = (64 << 16);                                    // tensor_dim1 = 64
  g1[3] = (32 << 16);                                    // tile_dim0 = 32
  g1[4] = 64;                                            // tile_dim1 = 64, tile_dim2 = 0
  g1[5] = (int)strideElems;                              // tensor_dim0_stride low32
  g1[6] = 0;
  g1[7] = 0;
  i32x4 zz4 = {0, 0, 0, 0};
  i32x8 zz8 = {0, 0, 0, 0, 0, 0, 0, 0};
  // 6-arg form (clang-23 / therock HIP headers): (g0, g1, g2, g3, g4, cpol)
  __builtin_amdgcn_tensor_load_to_lds(g0, g1, zz4, zz4, zz8, 0);
}
#endif

// ---------------------------------------------------------------------------
// Generic f16 WMMA GEMM: C[M,N](f32) = A[M,K](f16,row-major) @ BT[Npad,K](f16)
// + bias(f32,len N), act: 0=none 1=tanh 2=relu.  M%64==0, K%32==0, BT rows
// padded to a multiple of 64 (loads unguarded; stores guarded by n<N).
// Block 256 thr = 8 waves; tile 64x64; double-buffered LDS fed by the TDM.
// ---------------------------------------------------------------------------
#define LDT 40
__global__ __launch_bounds__(256) void gemm_wmma(
    const _Float16* __restrict__ A, const _Float16* __restrict__ BT,
    float* __restrict__ C, int M, int N, int K, int ldc,
    const float* __restrict__ bias, int act)
{
  __shared__ __align__(16) _Float16 As[2][64 * LDT];
  __shared__ __align__(16) _Float16 Bs[2][64 * LDT];
  const int tid = threadIdx.x, w = tid >> 5, lane = tid & 31;
  const int m0 = blockIdx.x * 64, n0 = blockIdx.y * 64;
  const int mi = w & 3, nj0 = (w >> 2) * 2;
  const int nk = K >> 5;
  v8f acc[2]; acc[0] = {}; acc[1] = {};

#if USE_TDM
  const _Float16* Ab = A + (size_t)m0 * K;
  const _Float16* Bb = BT + (size_t)n0 * K;
  if (w == 0) {
    tdm_load_tile_64x32(Ab, (uint32_t)(uintptr_t)&As[0][0], (uint32_t)K);
    tdm_load_tile_64x32(Bb, (uint32_t)(uintptr_t)&Bs[0][0], (uint32_t)K);
  }
#else
  const int lrow = tid >> 2, lcol = (tid & 3) * 8;
#endif

  for (int ks = 0; ks < nk; ++ks) {
    const int cur = ks & 1;
#if USE_TDM
    if (ks + 1 < nk) {
      if (w == 0) {
        tdm_load_tile_64x32(Ab + (ks + 1) * 32, (uint32_t)(uintptr_t)&As[cur ^ 1][0], (uint32_t)K);
        tdm_load_tile_64x32(Bb + (ks + 1) * 32, (uint32_t)(uintptr_t)&Bs[cur ^ 1][0], (uint32_t)K);
      }
      __builtin_amdgcn_s_wait_tensorcnt(2);   // current pair landed
    } else {
      __builtin_amdgcn_s_wait_tensorcnt(0);   // drain
    }
    __syncthreads();
#else
    *(v8h*)&As[cur][lrow * LDT + lcol] = *(const v8h*)&A[(size_t)(m0 + lrow) * K + ks * 32 + lcol];
    *(v8h*)&Bs[cur][lrow * LDT + lcol] = *(const v8h*)&BT[(size_t)(n0 + lrow) * K + ks * 32 + lcol];
    __syncthreads();
#endif
    v16h a  = load_frag_a(&As[cur][mi * 16 * LDT], LDT, lane);
    v16h b0 = load_frag_b(&Bs[cur][(nj0 + 0) * 16 * LDT], LDT, lane);
    v16h b1 = load_frag_b(&Bs[cur][(nj0 + 1) * 16 * LDT], LDT, lane);
    acc[0] = WMMA16(a, b0, acc[0]);
    acc[1] = WMMA16(a, b1, acc[1]);
    __syncthreads();                          // buffer reuse safe
  }

  const int nloc = lane & 15, hi = lane >> 4;
#pragma unroll
  for (int j = 0; j < 2; ++j) {
    int n = n0 + (nj0 + j) * 16 + nloc;
    if (n < N) {
      float bvv = bias ? bias[n] : 0.f;
#pragma unroll
      for (int r = 0; r < 8; ++r) {
        int m = m0 + mi * 16 + hi * 8 + r;
        float v = acc[j][r] + bvv;
        if (act == 1) v = tanhf(v);
        else if (act == 2) v = fmaxf(v, 0.f);
        C[(size_t)m * ldc + n] = v;
      }
    }
  }
}

// ---------------------------------------------------------------------------
// Persistent bidirectional LSTM. grid=(2) dirs, 1024 threads = 32 waves.
// Dyn LDS: hbuf f16[64][264] + Gbuf f32[64][1032]  (~298KB of 320KB WGP LDS).
// Per step: G = h @ WhT (WMMA, weights streamed from L2) + xp + b; gates in regs.
// ---------------------------------------------------------------------------
__global__ __launch_bounds__(1024) void lstm_kernel(
    const float* __restrict__ xp_f, const float* __restrict__ xp_b,
    const _Float16* __restrict__ whfT, const _Float16* __restrict__ whbT,
    const float* __restrict__ bf, const float* __restrict__ bb,
    float* __restrict__ facts)
{
  extern __shared__ char smem[];
  _Float16* hbuf = (_Float16*)smem;                 // 64 x 264
  float*    Gbuf = (float*)(smem + 64 * 264 * 2);   // 64 x 1032
  const int dir = blockIdx.x;
  const float*    xp   = dir ? xp_b : xp_f;
  const _Float16* whT  = dir ? whbT : whfT;
  const float*    bias = dir ? bb : bf;
  const int tid = threadIdx.x, w = tid >> 5, lane = tid & 31;
  const int mi = w & 3, njB = w >> 2;               // njB: 0..7
  const int nloc = lane & 15, hi = lane >> 4;

  for (int i = tid; i < 64 * 264; i += 1024) hbuf[i] = (_Float16)0.f;
  float c[16];
#pragma unroll
  for (int r = 0; r < 16; ++r) c[r] = 0.f;
  __syncthreads();

  for (int t = 0; t < NT; ++t) {
    const int tt = dir ? (NT - 1 - t) : t;
    v8f acc[8];
#pragma unroll
    for (int jj = 0; jj < 8; ++jj) acc[jj] = {};
#pragma unroll
    for (int kc = 0; kc < 8; ++kc) {                // K = 256
      v16h a = load_frag_a(&hbuf[mi * 16 * 264 + kc * 32], 264, lane);
#pragma unroll
      for (int jj = 0; jj < 8; ++jj) {
        int nt = njB + 8 * jj;                      // N-tile 0..63
        const _Float16* bp = &whT[(size_t)(nt * 16 + nloc) * 256 + kc * 32 + hi * 16];
        v16h b;
        *(v8h*)&b       = *(const v8h*)bp;
        *((v8h*)&b + 1) = *(const v8h*)(bp + 8);
        acc[jj] = WMMA16(a, b, acc[jj]);
      }
    }
#pragma unroll
    for (int jj = 0; jj < 8; ++jj) {
      int n = (njB + 8 * jj) * 16 + nloc;
      float bv = bias[n];
#pragma unroll
      for (int r = 0; r < 8; ++r) {
        int bidx = mi * 16 + hi * 8 + r;
        Gbuf[bidx * 1032 + n] = acc[jj][r] + bv + xp[((size_t)bidx * NT + tt) * NH4 + n];
      }
    }
    __syncthreads();
#pragma unroll
    for (int r = 0; r < 16; ++r) {
      int e = tid * 16 + r;
      int bb2 = e >> 8, j = e & 255;
      float gi = Gbuf[bb2 * 1032 + j];
      float gf = Gbuf[bb2 * 1032 + j + 256];
      float gg = Gbuf[bb2 * 1032 + j + 512];
      float go = Gbuf[bb2 * 1032 + j + 768];
      float cv = sigf(gf) * c[r] + sigf(gi) * tanhf(gg);
      c[r] = cv;
      float hv = sigf(go) * tanhf(cv);
      hbuf[bb2 * 264 + j] = (_Float16)hv;
      facts[((size_t)bb2 * NT + tt) * ND + dir * 256 + j] = hv;
    }
    __syncthreads();
  }
}

// ---------------------------------------------------------------------------
// Persistent attention-GRU. 1 block, 1024 thr = 32 waves, h in LDS f16[64][528].
// Per step both h@Ur and h@Uc via dual WMMA accumulators; gates fused in regs.
// ---------------------------------------------------------------------------
__global__ __launch_bounds__(1024) void attngru_kernel(
    const float* __restrict__ xr, const float* __restrict__ xc,
    const float* __restrict__ attn,
    const _Float16* __restrict__ urT, const _Float16* __restrict__ ucT,
    float* __restrict__ ep)
{
  extern __shared__ char smem[];
  _Float16* hbuf = (_Float16*)smem;                 // 64 x 528
  const int tid = threadIdx.x, w = tid >> 5, lane = tid & 31;
  const int mi = w & 3, njB = w >> 2;               // njB: 0..7
  const int nloc = lane & 15, hi = lane >> 4;
  for (int i = tid; i < 64 * 528; i += 1024) hbuf[i] = (_Float16)0.f;
  __syncthreads();

  for (int t = 0; t < NT; ++t) {
    v8f accR[4], accC[4];
#pragma unroll
    for (int jj = 0; jj < 4; ++jj) { accR[jj] = {}; accC[jj] = {}; }
#pragma unroll 4
    for (int kc = 0; kc < 16; ++kc) {               // K = 512
      v16h a = load_frag_a(&hbuf[mi * 16 * 528 + kc * 32], 528, lane);
#pragma unroll
      for (int jj = 0; jj < 4; ++jj) {
        int nt = njB + 8 * jj;                      // N-tile 0..31
        size_t boff = (size_t)(nt * 16 + nloc) * ND + kc * 32 + hi * 16;
        v16h br, bc;
        *(v8h*)&br       = *(const v8h*)&urT[boff];
        *((v8h*)&br + 1) = *(const v8h*)&urT[boff + 8];
        *(v8h*)&bc       = *(const v8h*)&ucT[boff];
        *((v8h*)&bc + 1) = *(const v8h*)&ucT[boff + 8];
        accR[jj] = WMMA16(a, br, accR[jj]);
        accC[jj] = WMMA16(a, bc, accC[jj]);
      }
    }
    __syncthreads();                                // all h reads done
#pragma unroll
    for (int jj = 0; jj < 4; ++jj) {
      int j = (njB + 8 * jj) * 16 + nloc;
#pragma unroll
      for (int r = 0; r < 8; ++r) {
        int b = mi * 16 + hi * 8 + r;
        size_t rowoff = ((size_t)b * NT + t) * ND + j;
        float rv  = sigf(xr[rowoff] + accR[jj][r]);       // xr includes br
        float hcv = tanhf(xc[rowoff] + rv * accC[jj][r]); // xc includes bc
        float gt  = attn[b * NT + t];
        float hold = (float)hbuf[b * 528 + j];
        float hn = gt * hcv + (1.f - gt) * hold;
        hbuf[b * 528 + j] = (_Float16)hn;
        if (t == NT - 1) ep[(size_t)b * ND + j] = hn;
      }
    }
    __syncthreads();
  }
}

// --------------------------- small helper kernels ---------------------------
__global__ void embed_pack(const int* __restrict__ tok, const float* __restrict__ emb,
                           _Float16* __restrict__ xh)
{
  size_t total = (size_t)NR * NEP;
  for (size_t i = (size_t)blockIdx.x * blockDim.x + threadIdx.x; i < total;
       i += (size_t)gridDim.x * blockDim.x) {
    size_t row = i / NEP; int k = (int)(i % NEP);
    float v = (k < NE) ? emb[(size_t)tok[row] * NE + k] : 0.f;
    xh[i] = (_Float16)v;
  }
}

// W[K,N] (f32 row-major) -> WT[Np,Kp] (f16, N-major, zero padded)
__global__ void pack_wT(const float* __restrict__ W, _Float16* __restrict__ WT,
                        int K, int N, int Kp, int Np)
{
  size_t total = (size_t)Np * Kp;
  for (size_t i = (size_t)blockIdx.x * blockDim.x + threadIdx.x; i < total;
       i += (size_t)gridDim.x * blockDim.x) {
    int n = (int)(i / Kp), k = (int)(i % Kp);
    float v = (n < N && k < K) ? W[(size_t)k * N + n] : 0.f;
    WT[i] = (_Float16)v;
  }
}

__global__ void cvt_f16(const float* __restrict__ s, _Float16* __restrict__ d, size_t n)
{
  for (size_t i = (size_t)blockIdx.x * blockDim.x + threadIdx.x; i < n;
       i += (size_t)gridDim.x * blockDim.x) d[i] = (_Float16)s[i];
}

__global__ void copy_f32(const float* __restrict__ s, float* __restrict__ d, int n)
{
  int i = blockIdx.x * blockDim.x + threadIdx.x;
  if (i < n) d[i] = s[i];
}

__global__ void build_z(const float* __restrict__ facts, const float* __restrict__ q,
                        const float* __restrict__ m, _Float16* __restrict__ z)
{
  size_t total = (size_t)NR * ND;
  for (size_t i = (size_t)blockIdx.x * blockDim.x + threadIdx.x; i < total;
       i += (size_t)gridDim.x * blockDim.x) {
    size_t row = i >> 9; int j = (int)(i & 511);
    int b = (int)(row / NT);
    float f = facts[i];
    float qv = q[b * ND + j], mv = m[b * ND + j];
    _Float16* zp = z + row * ND4;
    zp[j]        = (_Float16)(f * qv);
    zp[512 + j]  = (_Float16)(f * mv);
    zp[1024 + j] = (_Float16)fabsf(f - qv);
    zp[1536 + j] = (_Float16)fabsf(f - mv);
  }
}

__global__ void attn_score(const float* __restrict__ hatt, const float* __restrict__ W2,
                           const float* __restrict__ b2, const int* __restrict__ len,
                           float* __restrict__ attn)
{
  __shared__ float red[128];
  int b = blockIdx.x, tid = threadIdx.x;
  bool has = tid < NT;
  float sv = -1e30f;
  if (has) {
    float dot = 0.f;
    const float* hp = &hatt[((size_t)b * NT + tid) * NE];
    for (int e = 0; e < NE; ++e) dot += hp[e] * W2[e];
    sv = dot + b2[0];
    if (tid >= len[b]) sv = -1e9f;
  }
  red[tid] = sv; __syncthreads();
  for (int off = 64; off > 0; off >>= 1) {
    if (tid < off) red[tid] = fmaxf(red[tid], red[tid + off]);
    __syncthreads();
  }
  float mx = red[0]; __syncthreads();
  float ex = has ? __expf(sv - mx) : 0.f;
  red[tid] = ex; __syncthreads();
  for (int off = 64; off > 0; off >>= 1) {
    if (tid < off) red[tid] += red[tid + off];
    __syncthreads();
  }
  if (has) attn[(size_t)b * NT + tid] = ex / red[0];
}

__global__ void build_u(const float* __restrict__ m, const float* __restrict__ ep,
                        const float* __restrict__ q, _Float16* __restrict__ u)
{
  int i = blockIdx.x * blockDim.x + threadIdx.x;
  if (i >= NB * ND3) return;
  int b = i / ND3, j = i % ND3;
  float v = (j < 512) ? m[b * ND + j]
          : (j < 1024) ? ep[b * ND + (j - 512)]
          : q[b * ND + (j - 1024)];
  u[i] = (_Float16)v;
}

__global__ void final_out(const float* __restrict__ m, const float* __restrict__ q,
                          const float* __restrict__ Wo, const float* __restrict__ bo,
                          float* __restrict__ out)
{
  int b = threadIdx.x;
  if (b >= NB) return;
  float acc = bo[0];
  for (int j = 0; j < ND; ++j) acc += m[b * ND + j] * Wo[j];
  for (int j = 0; j < ND; ++j) acc += q[b * ND + j] * Wo[ND + j];
  out[b] = 1.f / (1.f + __expf(-acc));
}

// ---------------------------------------------------------------------------
extern "C" void kernel_launch(void* const* d_in, const int* in_sizes, int n_in,
                              void* d_out, int out_size, void* d_ws, size_t ws_size,
                              hipStream_t stream) {
  const int*   tokens  = (const int*)  d_in[0];
  const int*   lengths = (const int*)  d_in[1];
  const float* emb     = (const float*)d_in[2];
  const float* Wx_f    = (const float*)d_in[3];
  const float* Wh_f    = (const float*)d_in[4];
  const float* b_f     = (const float*)d_in[5];
  const float* Wx_b    = (const float*)d_in[6];
  const float* Wh_b    = (const float*)d_in[7];
  const float* b_b     = (const float*)d_in[8];
  const float* W1      = (const float*)d_in[9];
  const float* b1      = (const float*)d_in[10];
  const float* W2      = (const float*)d_in[11];
  const float* b2      = (const float*)d_in[12];
  const float* Wr      = (const float*)d_in[13];
  const float* Ur      = (const float*)d_in[14];
  const float* br      = (const float*)d_in[15];
  const float* Wc      = (const float*)d_in[16];
  const float* Uc      = (const float*)d_in[17];
  const float* bc      = (const float*)d_in[18];
  const float* q       = (const float*)d_in[19];
  const float* W_hops  = (const float*)d_in[20];
  const float* b_hops  = (const float*)d_in[21];
  const float* Wo      = (const float*)d_in[22];
  const float* bo      = (const float*)d_in[23];

  char* base = (char*)d_ws;
  size_t off = 0;
  auto take = [&](size_t bytes) -> char* {
    char* r = base + off;
    off = (off + bytes + 255) & ~(size_t)255;
    return r;
  };
  _Float16* x_h    = (_Float16*)take((size_t)NR * NEP * 2);
  _Float16* wxfT   = (_Float16*)take((size_t)NH4 * NEP * 2);
  _Float16* wxbT   = (_Float16*)take((size_t)NH4 * NEP * 2);
  _Float16* whfT   = (_Float16*)take((size_t)NH4 * 256 * 2);
  _Float16* whbT   = (_Float16*)take((size_t)NH4 * 256 * 2);
  _Float16* w1T    = (_Float16*)take((size_t)320 * ND4 * 2);   // 320 rows (N padded)
  _Float16* wrT    = (_Float16*)take((size_t)ND * ND * 2);
  _Float16* wcT    = (_Float16*)take((size_t)ND * ND * 2);
  _Float16* urT    = (_Float16*)take((size_t)ND * ND * 2);
  _Float16* ucT    = (_Float16*)take((size_t)ND * ND * 2);
  _Float16* whopT  = (_Float16*)take((size_t)3 * ND * ND3 * 2);
  float*    xp_f   = (float*)take((size_t)NR * NH4 * 4);   // reused as z (f16, same bytes)
  float*    xp_b   = (float*)take((size_t)NR * NH4 * 4);   // reused as xr|xc
  float*    facts  = (float*)take((size_t)NR * ND * 4);
  _Float16* factsh = (_Float16*)take((size_t)NR * ND * 2);
  float*    hatt   = (float*)take((size_t)NR * NE * 4);
  float*    attn   = (float*)take((size_t)NR * 4);
  float*    mbuf   = (float*)take((size_t)NB * ND * 4);
  float*    epbuf  = (float*)take((size_t)NB * ND * 4);
  _Float16* u_h    = (_Float16*)take((size_t)NB * ND3 * 2);

  _Float16* z  = (_Float16*)xp_f;                 // NR x 2048 f16
  float*    xr = xp_b;                            // NR x 512 f32
  float*    xc = xp_b + (size_t)NR * ND;          // NR x 512 f32

  // Pack embedding rows + weights to f16 (N-major, rows padded to mult of 64).
  embed_pack<<<2048, 256, 0, stream>>>(tokens, emb, x_h);
  pack_wT<<<1024, 256, 0, stream>>>(Wx_f, wxfT, NE, NH4, NEP, NH4);
  pack_wT<<<1024, 256, 0, stream>>>(Wx_b, wxbT, NE, NH4, NEP, NH4);
  pack_wT<<<1024, 256, 0, stream>>>(Wh_f, whfT, 256, NH4, 256, NH4);
  pack_wT<<<1024, 256, 0, stream>>>(Wh_b, whbT, 256, NH4, 256, NH4);
  pack_wT<<<1024, 256, 0, stream>>>(W1,   w1T,  ND4, NE, ND4, 320);
  pack_wT<<<1024, 256, 0, stream>>>(Wr,   wrT,  ND, ND, ND, ND);
  pack_wT<<<1024, 256, 0, stream>>>(Wc,   wcT,  ND, ND, ND, ND);
  pack_wT<<<1024, 256, 0, stream>>>(Ur,   urT,  ND, ND, ND, ND);
  pack_wT<<<1024, 256, 0, stream>>>(Uc,   ucT,  ND, ND, ND, ND);
  for (int i = 0; i < 3; ++i)
    pack_wT<<<1024, 256, 0, stream>>>(W_hops + (size_t)i * ND3 * ND,
                                      whopT + (size_t)i * ND * ND3, ND3, ND, ND3, ND);

  // LSTM input projections: xp = x @ Wx + b   (7744x320 @ 320x1024)
  gemm_wmma<<<dim3(NR / 64, NH4 / 64), 256, 0, stream>>>(x_h, wxfT, xp_f, NR, NH4, NEP, NH4, b_f, 0);
  gemm_wmma<<<dim3(NR / 64, NH4 / 64), 256, 0, stream>>>(x_h, wxbT, xp_b, NR, NH4, NEP, NH4, b_b, 0);

  // Persistent BiLSTM (2 WGPs, 32 waves each, 298KB LDS)
  size_t lstm_smem = (size_t)64 * 264 * 2 + (size_t)64 * 1032 * 4;
  lstm_kernel<<<2, 1024, lstm_smem, stream>>>(xp_f, xp_b, whfT, whbT, b_f, b_b, facts);

  cvt_f16<<<4096, 256, 0, stream>>>(facts, factsh, (size_t)NR * ND);

  // Hop-invariant GRU input projections (include biases): xr=facts@Wr+br, xc=facts@Wc+bc
  gemm_wmma<<<dim3(NR / 64, ND / 64), 256, 0, stream>>>(factsh, wrT, xr, NR, ND, ND, ND, br, 0);
  gemm_wmma<<<dim3(NR / 64, ND / 64), 256, 0, stream>>>(factsh, wcT, xc, NR, ND, ND, ND, bc, 0);

  // m = q
  copy_f32<<<(NB * ND + 255) / 256, 256, 0, stream>>>(q, mbuf, NB * ND);

  size_t gru_smem = (size_t)64 * 528 * 2;
  for (int hop = 0; hop < 3; ++hop) {
    build_z<<<8192, 256, 0, stream>>>(facts, q, mbuf, z);
    gemm_wmma<<<dim3(NR / 64, 5), 256, 0, stream>>>(z, w1T, hatt, NR, NE, ND4, NE, b1, 1);
    attn_score<<<NB, 128, 0, stream>>>(hatt, W2, b2, lengths, attn);
    attngru_kernel<<<1, 1024, gru_smem, stream>>>(xr, xc, attn, urT, ucT, epbuf);
    build_u<<<(NB * ND3 + 255) / 256, 256, 0, stream>>>(mbuf, epbuf, q, u_h);
    gemm_wmma<<<dim3(1, ND / 64), 256, 0, stream>>>(u_h, whopT + (size_t)hop * ND * ND3,
                                                    mbuf, NB, ND, ND3, ND,
                                                    b_hops + (size_t)hop * ND, 2);
  }

  final_out<<<1, 64, 0, stream>>>(mbuf, q, Wo, bo, (float*)d_out);
}
